// TCNModel_73048803770559
// MI455X (gfx1250) — compile-verified, matrix operand order checked by reading, not run on hardware
//
#include <hip/hip_runtime.h>
#include <hip/hip_bf16.h>

#define T_LEN 512
#define INCH  128
#define HID   256

typedef __attribute__((ext_vector_type(16))) __bf16 v16bf;
typedef __attribute__((ext_vector_type(8)))  float  v8f;

// K-permutation making each half-lane's WMMA fragment contiguous in LDS:
// order [0..7, 16..23, 8..15, 24..31]. Involutive (perm == perm^-1).
__device__ __forceinline__ int permk(int j) {
  return (((j >> 3) & 1) << 4) + (((j >> 4) & 1) << 3) + (j & 7);
}

// ---------------------------------------------------------------------------
// Transpose + bf16 convert: x [320, T, 128] f32 -> subj [64,128,T], neigh
// [256,128,T].  LDS-tiled so both global reads and writes are coalesced.
// ---------------------------------------------------------------------------
__global__ void k_transpose_split(const float* __restrict__ x,
                                  __bf16* __restrict__ subj,
                                  __bf16* __restrict__ neigh) {
  __shared__ float sh[32][INCH + 1];   // +1 pad: conflict-free column reads
  const int n  = blockIdx.x;   // 0..319
  const int tc = blockIdx.y;   // 0..15, 32 timesteps each
  const int b = n / 5, g = n % 5;
  __bf16* dst = (g == 0) ? (subj + (size_t)b * INCH * T_LEN)
                         : (neigh + (size_t)(b * 4 + g - 1) * INCH * T_LEN);
  const float* src = x + (size_t)n * T_LEN * INCH + (size_t)tc * 32 * INCH;
  for (int idx = threadIdx.x; idx < 32 * INCH; idx += blockDim.x) {
    int t = idx >> 7, c = idx & 127;         // coalesced read over c
    sh[t][c] = src[t * INCH + c];
  }
  __syncthreads();
  for (int idx = threadIdx.x; idx < 32 * INCH; idx += blockDim.x) {
    int c = idx >> 5, t = idx & 31;          // coalesced write over t
    dst[c * T_LEN + tc * 32 + t] = (__bf16)sh[t][c];
  }
}

// ---------------------------------------------------------------------------
// Dilated causal conv as WMMA GEMM.
//  in  : bf16 [Ns, Cin, T]
//  W   : f32  [256, Cin, TAPS]  (converted to bf16 while staging to LDS)
//  out : bf16 [Ns, 256, T]
// Block tile: M=128 x N=128, 8 waves, each wave 16x128 (8 wmma accumulators).
// K-loop over 32-channel chunks; TAPS accumulated via time-shifted B tiles.
// LDS layouts are K-permuted so each A/B fragment is a single 32-byte read.
// Epilogue: +bias, relu, optional residual add (+relu again), exact to ref.
// ---------------------------------------------------------------------------
template <int TAPS>
__global__ __launch_bounds__(256)
void k_conv_wmma(const __bf16* __restrict__ in, int Cin,
                 const float* __restrict__ W, const float* __restrict__ bias,
                 int dil, int do_relu,
                 const __bf16* __restrict__ addsrc,
                 __bf16* __restrict__ out) {
  __shared__ __align__(32) __bf16 sW[TAPS * 128 * 32];  // [tap][co][perm-k]
  __shared__ __align__(32) __bf16 sXt[144 * 32];        // [t halo16+128][perm-k]

  const int seq   = blockIdx.x;
  const int mbase = blockIdx.y * 128;
  const int t0    = blockIdx.z * 128;
  const int tid   = threadIdx.x;
  const int lane  = tid & 31;
  const int wave  = tid >> 5;

  const __bf16* inS = in + (size_t)seq * Cin * T_LEN;

  v8f acc[8] = {};

  const int chunks = Cin >> 5;
  const int m   = lane & 15;
  const int kh2 = (lane >> 4) << 4;  // element offset of this half-lane's frag

  for (int ch = 0; ch < chunks; ++ch) {
    // ---- stage weights (f32 -> bf16), K-permuted; global reads coalesced ----
    for (int idx = tid; idx < 128 * 32 * TAPS; idx += 256) {
      int co  = idx / (32 * TAPS);
      int rem = idx - co * 32 * TAPS;
      int ci  = rem / TAPS;
      int tp  = rem - ci * TAPS;
      sW[tp * 4096 + co * 32 + permk(ci)] =
          (__bf16)W[(size_t)(mbase + co) * (Cin * TAPS) + (ch * 32 + ci) * TAPS + tp];
    }
    // ---- stage input slab transposed [t][perm-k], causal halo 16 zero-filled ----
    for (int idx = tid; idx < 32 * 144; idx += 256) {
      int r = idx / 144;         // channel within chunk
      int c = idx - r * 144;     // local time (halo included)
      int gt = t0 - 16 + c;
      __bf16 v = (__bf16)0.0f;
      if (gt >= 0 && gt < T_LEN) v = inS[(ch * 32 + r) * T_LEN + gt];
      sXt[c * 32 + permk(r)] = v;
    }
    // prefetch next chunk's input rows (global_prefetch_b8)
    if (ch + 1 < chunks)
      __builtin_prefetch(&inS[((ch + 1) * 32 + (tid >> 3)) * T_LEN + t0], 0, 0);
    __syncthreads();

#pragma unroll
    for (int tp = 0; tp < TAPS; ++tp) {
      // A fragment: one contiguous 32B read per lane, reused across 8 N-tiles
      v16bf a = *(const v16bf*)(sW + tp * 4096 + (wave * 16 + m) * 32 + kh2);
      const int cb = 16 - tp * dil + m;  // shifted column base for this tap
#pragma unroll
      for (int nt = 0; nt < 8; ++nt) {
        // B fragment: one contiguous 32B read per lane
        v16bf bfr = *(const v16bf*)(sXt + (cb + nt * 16) * 32 + kh2);
        acc[nt] = __builtin_amdgcn_wmma_f32_16x16x32_bf16(
            false, a, false, bfr, (short)0, acc[nt], false, false);
      }
    }
    __syncthreads();
  }

  // ---- epilogue: bias, relu, fused residual (+relu), store bf16 ----
  const int rbase = mbase + wave * 16 + ((lane >> 4) << 3);
  const int col0  = t0 + (lane & 15);
#pragma unroll
  for (int nt = 0; nt < 8; ++nt) {
    const int c = col0 + nt * 16;
#pragma unroll
    for (int j = 0; j < 8; ++j) {
      const int row = rbase + j;
      float v = acc[nt][j] + bias[row];
      if (do_relu) v = fmaxf(v, 0.0f);
      if (addsrc) {
        v += (float)addsrc[(size_t)seq * HID * T_LEN + row * T_LEN + c];
        if (do_relu) v = fmaxf(v, 0.0f);
      }
      out[(size_t)seq * HID * T_LEN + row * T_LEN + c] = (__bf16)v;
    }
  }
}

// ---------------------------------------------------------------------------
// Last-valid feature extraction: feat[s,h] = act[s,h,len-1] (0 if len==0)
// ---------------------------------------------------------------------------
__global__ void k_extract_last(const __bf16* __restrict__ act,
                               const int* __restrict__ lens,
                               float* __restrict__ feat, int subj) {
  const int s = blockIdx.x;
  const int li = subj ? (s * 5) : ((s >> 2) * 5 + 1 + (s & 3));
  const int l = lens[li];
  const int h = threadIdx.x;
  float v = 0.0f;
  if (l > 0) v = (float)act[(size_t)s * HID * T_LEN + h * T_LEN + (l - 1)];
  feat[s * HID + h] = v;
}

// ---------------------------------------------------------------------------
// Masked mean pool over {subject, 4 neighbors} -> hbuf[b, coloff + h]
// ---------------------------------------------------------------------------
__global__ void k_pool(const float* __restrict__ fs, const float* __restrict__ fn,
                       const int* __restrict__ lens, float* __restrict__ hbuf,
                       int coloff) {
  const int b = blockIdx.x;
  const int h = threadIdx.x;
  float acc = fs[b * HID + h];
  float cnt = 1.0f;
  for (int g = 0; g < 4; ++g) {
    if (lens[b * 5 + 1 + g] > 0) {
      acc += fn[(b * 4 + g) * HID + h];
      cnt += 1.0f;
    }
  }
  hbuf[b * (2 * HID) + coloff + h] = acc / cnt;
}

// ---------------------------------------------------------------------------
// Heads: hid = leaky_relu(h @ Wf + bf);  out = hid @ Wo + bo  (tiny, scalar)
// ---------------------------------------------------------------------------
__global__ void k_head_fc(const float* __restrict__ hbuf,
                          const float* w0, const float* c0,
                          const float* w1, const float* c1,
                          const float* w2, const float* c2,
                          float* __restrict__ hid) {
  __shared__ float sh[2 * HID];
  const int head = blockIdx.x, b = blockIdx.y, j = threadIdx.x;
  const float* w  = head == 0 ? w0 : (head == 1 ? w1 : w2);
  const float* bb = head == 0 ? c0 : (head == 1 ? c1 : c2);
  for (int i = threadIdx.x; i < 2 * HID; i += blockDim.x)
    sh[i] = hbuf[b * 2 * HID + i];
  __syncthreads();
  float a = bb[j];
  for (int i = 0; i < 2 * HID; ++i) a += sh[i] * w[i * HID + j];
  hid[((size_t)head * 64 + b) * HID + j] = a > 0.0f ? a : 0.01f * a;
}

__global__ void k_head_out(const float* __restrict__ hid,
                           const float* w0, const float* c0,
                           const float* w1, const float* c1,
                           const float* w2, const float* c2,
                           float* __restrict__ outp) {
  const int head = blockIdx.x, b = threadIdx.x;
  const float* w  = head == 0 ? w0 : (head == 1 ? w1 : w2);
  const float* bb = head == 0 ? c0 : (head == 1 ? c1 : c2);
  const float* hr = hid + ((size_t)head * 64 + b) * HID;
  float a = bb[0];
  for (int j = 0; j < HID; ++j) a += hr[j] * w[j];
  outp[head * 64 + b] = a;
}

// ---------------------------------------------------------------------------
// Host orchestration
// ---------------------------------------------------------------------------
struct Tcn {
  const float *w1[4], *b1[4], *w2[4], *b2[4], *wd, *bd;
};

static Tcn grab_tcn(void* const* d, int i) {
  Tcn t;
  t.w1[0] = (const float*)d[i + 0]; t.b1[0] = (const float*)d[i + 1];
  t.w2[0] = (const float*)d[i + 2]; t.b2[0] = (const float*)d[i + 3];
  t.wd    = (const float*)d[i + 4]; t.bd    = (const float*)d[i + 5];
  int k = i + 6;
  for (int b = 1; b < 4; ++b) {
    t.w1[b] = (const float*)d[k++]; t.b1[b] = (const float*)d[k++];
    t.w2[b] = (const float*)d[k++]; t.b2[b] = (const float*)d[k++];
  }
  return t;
}

static __bf16* run_tcn(const __bf16* xin, int Ns, const Tcn& t,
                       __bf16* b0, __bf16* b1, __bf16* b2, hipStream_t stream) {
  dim3 blk(256);
  auto conv3 = [&](const __bf16* in, int Cin, const float* W, const float* B,
                   int dil, int relu, const __bf16* add, __bf16* out) {
    dim3 grid(Ns, 2, 4);
    k_conv_wmma<3><<<grid, blk, 0, stream>>>(in, Cin, W, B, dil, relu, add, out);
  };
  auto conv1 = [&](const __bf16* in, int Cin, const float* W, const float* B,
                   __bf16* out) {
    dim3 grid(Ns, 2, 4);
    k_conv_wmma<1><<<grid, blk, 0, stream>>>(in, Cin, W, B, 1, 0, nullptr, out);
  };
  // block 0: downsample residual + conv1 + conv2(+res)
  conv1(xin, INCH, t.wd, t.bd, b2);
  conv3(xin, INCH, t.w1[0], t.b1[0], 1, 1, nullptr, b0);
  conv3(b0, HID, t.w2[0], t.b2[0], 1, 1, b2, b1);
  __bf16* cur = b1;
  __bf16* oth = b2;
  for (int i = 1; i < 4; ++i) {
    int d = 1 << i;
    conv3(cur, HID, t.w1[i], t.b1[i], d, 1, nullptr, b0);
    conv3(b0, HID, t.w2[i], t.b2[i], d, 1, cur, oth);
    __bf16* tmp = cur; cur = oth; oth = tmp;
  }
  return cur;
}

extern "C" void kernel_launch(void* const* d_in, const int* in_sizes, int n_in,
                              void* d_out, int out_size, void* d_ws, size_t ws_size,
                              hipStream_t stream) {
  (void)in_sizes; (void)n_in; (void)out_size; (void)ws_size;

  const float* x_acc    = (const float*)d_in[0];
  const float* x_mic    = (const float*)d_in[1];
  const int*   acc_lens = (const int*)d_in[2];
  const int*   mic_lens = (const int*)d_in[3];
  Tcn ts_acc = grab_tcn(d_in, 4);
  Tcn tn_acc = grab_tcn(d_in, 22);
  Tcn ts_mic = grab_tcn(d_in, 40);
  Tcn tn_mic = grab_tcn(d_in, 58);
  const float* fcw[3] = {(const float*)d_in[76], (const float*)d_in[80], (const float*)d_in[84]};
  const float* fcb[3] = {(const float*)d_in[77], (const float*)d_in[81], (const float*)d_in[85]};
  const float* ow[3]  = {(const float*)d_in[78], (const float*)d_in[82], (const float*)d_in[86]};
  const float* ob[3]  = {(const float*)d_in[79], (const float*)d_in[83], (const float*)d_in[87]};

  char* ws = (char*)d_ws;
  size_t off = 0;
  auto alloc = [&](size_t bytes) {
    char* p = ws + off;
    off = (off + bytes + 255) & ~(size_t)255;
    return p;
  };
  __bf16* xsub  = (__bf16*)alloc((size_t)64  * INCH * T_LEN * 2);
  __bf16* xnei  = (__bf16*)alloc((size_t)256 * INCH * T_LEN * 2);
  __bf16* b0    = (__bf16*)alloc((size_t)256 * HID * T_LEN * 2);
  __bf16* b1    = (__bf16*)alloc((size_t)256 * HID * T_LEN * 2);
  __bf16* b2    = (__bf16*)alloc((size_t)256 * HID * T_LEN * 2);
  float* feat_s = (float*)alloc((size_t)64 * HID * 4);
  float* feat_n = (float*)alloc((size_t)256 * HID * 4);
  float* hbuf   = (float*)alloc((size_t)64 * 2 * HID * 4);
  float* hid3   = (float*)alloc((size_t)3 * 64 * HID * 4);

  auto modality = [&](const float* x, const int* lens, const Tcn& ts, const Tcn& tn,
                      int coloff) {
    k_transpose_split<<<dim3(320, 16), 256, 0, stream>>>(x, xsub, xnei);
    __bf16* sa = run_tcn(xsub, 64, ts, b0, b1, b2, stream);
    k_extract_last<<<64, HID, 0, stream>>>(sa, lens, feat_s, 1);
    __bf16* na = run_tcn(xnei, 256, tn, b0, b1, b2, stream);
    k_extract_last<<<256, HID, 0, stream>>>(na, lens, feat_n, 0);
    k_pool<<<64, HID, 0, stream>>>(feat_s, feat_n, lens, hbuf, coloff);
  };
  modality(x_acc, acc_lens, ts_acc, tn_acc, 0);
  modality(x_mic, mic_lens, ts_mic, tn_mic, HID);

  k_head_fc<<<dim3(3, 64), HID, 0, stream>>>(hbuf, fcw[0], fcb[0], fcw[1], fcb[1],
                                             fcw[2], fcb[2], hid3);
  k_head_out<<<3, 64, 0, stream>>>(hid3, ow[0], ob[0], ow[1], ob[1], ow[2], ob[2],
                                   (float*)d_out);
}
